// LengthRegulator_87205015978520
// MI455X (gfx1250) — matrix operand order repaired and checked by reading, not compiled, not run on hardware
//
#include <hip/hip_runtime.h>
#include <hip/hip_bf16.h>

typedef __attribute__((ext_vector_type(16))) _Float16 v16h;
typedef __attribute__((ext_vector_type(8)))  _Float16 v8h;
typedef __attribute__((ext_vector_type(8)))  float    v8f;

constexpr int BN = 32;          // batch
constexpr int TI = 1024;        // T_in
constexpr int TP = TI + 10;     // time-padded (pad 5 each side)
constexpr int HN = 384;         // hidden
constexpr int KW = 11;          // conv taps
constexpr int KD = HN * KW;     // GEMM reduction dim = 4224
constexpr int TO = 8192;        // T_out
constexpr float LN_EPS = 1e-5f;

// ---------------------------------------------------------------------------
// Packing kernels
// ---------------------------------------------------------------------------
__global__ __launch_bounds__(256) void pack_x_kernel(const float* __restrict__ x,
                                                     _Float16* __restrict__ xh) {
    int i = blockIdx.x * 256 + threadIdx.x;          // over BN*TI*HN
    if (i >= BN * TI * HN) return;
    int ch = i % HN;
    int r  = i / HN;
    int t  = r % TI;
    int b  = r / TI;
    xh[((size_t)b * TP + t + 5) * HN + ch] = (_Float16)x[i];
}

// conv weight (O, I, K) -> wp[o][tap*HN + i]  (B-columns contiguous in K-dim)
__global__ __launch_bounds__(256) void pack_w_kernel(const float* __restrict__ w,
                                                     _Float16* __restrict__ wp) {
    int i = blockIdx.x * 256 + threadIdx.x;          // over HN*HN*KW
    if (i >= HN * HN * KW) return;
    int k  = i % KW;
    int r  = i / KW;
    int ii = r % HN;
    int o  = r / HN;
    wp[(size_t)o * KD + k * HN + ii] = (_Float16)w[i];
}

// zero the 5 pad rows at each end (both padded f16 buffers), every call
__global__ __launch_bounds__(256) void zero_pads_kernel(_Float16* __restrict__ xh,
                                                        _Float16* __restrict__ mh) {
    int i = blockIdx.x * 256 + threadIdx.x;          // over 2*BN*10*HN
    const int per = BN * 10 * HN;
    if (i >= 2 * per) return;
    _Float16* buf = (i < per) ? xh : mh;
    int j  = i % per;
    int ch = j % HN;
    int r  = j / HN;
    int pr = r % 10;
    int b  = r / 10;
    int t  = (pr < 5) ? pr : (TP - 10 + pr);
    buf[((size_t)b * TP + t) * HN + ch] = (_Float16)0.f;
}

// ---------------------------------------------------------------------------
// Conv-as-GEMM with v_wmma_f32_16x16x32_f16, 2x2 register tiling per wave.
//   out[b,t,o] = sum_{tap,i} xh[b, t+tap, i] * wp[o, tap*HN+i]  (+ bias)
// Wave tile = 32M x 32N (4 accumulators); block = 4 waves (2x2) = 64M x 64N.
// grid = (BN*TI/64, HN/64), block = 128
// ---------------------------------------------------------------------------
__device__ __forceinline__ v16h mk_frag(v8h lo, v8h hi) {
    v16h r;
#pragma unroll
    for (int e = 0; e < 8; ++e) { r[e] = lo[e]; r[e + 8] = hi[e]; }
    return r;
}

__global__ __launch_bounds__(128) void conv_wmma_kernel(
        const _Float16* __restrict__ xh,   // (BN, TP, HN)
        const _Float16* __restrict__ wp,   // (HN, KD)
        const float*    __restrict__ bias, // (HN)
        float*          __restrict__ out)  // (BN, TI, HN)
{
    const int lane = threadIdx.x & 31;
    const int wave = threadIdx.x >> 5;
    const int h    = lane >> 4;                 // half-wave select
    const int mn   = lane & 15;                 // row/col within 16-tile
    const int bt   = blockIdx.x;                // M super-tile id
    const int b    = bt / (TI / 64);
    const int t0   = (bt % (TI / 64)) * 64 + (wave >> 1) * 32;  // wave M base
    const int o0   = blockIdx.y * 64 + (wave & 1) * 32;          // wave N base

    // A rows for the two 16-row subtiles this wave owns
    const _Float16* arow0 = xh + ((size_t)b * TP + (t0 + mn)) * HN;
    const _Float16* arow1 = arow0 + (size_t)16 * HN;
    // B columns for the two 16-col subtiles
    const _Float16* bcol0 = wp + (size_t)(o0 + mn) * KD;
    const _Float16* bcol1 = bcol0 + (size_t)16 * KD;

    v8f c00 = {}, c01 = {}, c10 = {}, c11 = {};
    for (int tap = 0; tap < KW; ++tap) {
        const size_t off = (size_t)tap * HN;
#pragma unroll
        for (int i0 = 0; i0 < HN; i0 += 32) {
            // A fragment (16x32): lane half h -> K chunks [8h..8h+7], [16+8h..+7]
            v16h A0 = mk_frag(*(const v8h*)(arow0 + off + i0 + 8 * h),
                              *(const v8h*)(arow0 + off + i0 + 16 + 8 * h));
            v16h A1 = mk_frag(*(const v8h*)(arow1 + off + i0 + 8 * h),
                              *(const v8h*)(arow1 + off + i0 + 16 + 8 * h));
            // B fragment (32x16): lane half h -> K chunks [8h..8h+7], [16+8h..+7] of its column
            v16h B0 = mk_frag(*(const v8h*)(bcol0 + off + i0 + 8 * h),
                              *(const v8h*)(bcol0 + off + i0 + 16 + 8 * h));
            v16h B1 = mk_frag(*(const v8h*)(bcol1 + off + i0 + 8 * h),
                              *(const v8h*)(bcol1 + off + i0 + 16 + 8 * h));
            c00 = __builtin_amdgcn_wmma_f32_16x16x32_f16(false, A0, false, B0,
                                                         (short)0, c00, false, false);
            c01 = __builtin_amdgcn_wmma_f32_16x16x32_f16(false, A0, false, B1,
                                                         (short)0, c01, false, false);
            c10 = __builtin_amdgcn_wmma_f32_16x16x32_f16(false, A1, false, B0,
                                                         (short)0, c10, false, false);
            c11 = __builtin_amdgcn_wmma_f32_16x16x32_f16(false, A1, false, B1,
                                                         (short)0, c11, false, false);
        }
    }
    // C/D layout: element j -> row M = j + 8*h, col N = lane&15
    const float bo0 = bias[o0 + mn];
    const float bo1 = bias[o0 + 16 + mn];
    float* obase = out + ((size_t)b * TI + t0) * HN + o0 + mn;
#pragma unroll
    for (int j = 0; j < 8; ++j) {
        const size_t r0 = (size_t)(j + 8 * h) * HN;
        const size_t r1 = (size_t)(j + 8 * h + 16) * HN;
        obase[r0]      = c00[j] + bo0;
        obase[r0 + 16] = c01[j] + bo1;
        obase[r1]      = c10[j] + bo0;
        obase[r1 + 16] = c11[j] + bo1;
    }
}

// ---------------------------------------------------------------------------
// LayerNorm(H) + ReLU -> padded f16 (input to next conv). Wave per row.
// ---------------------------------------------------------------------------
__global__ __launch_bounds__(256) void ln_relu_f16_kernel(
        const float* __restrict__ vin,    // (BN, TI, HN)
        const float* __restrict__ gamma,
        const float* __restrict__ beta,
        _Float16*    __restrict__ outp)   // (BN, TP, HN), row t -> t+5
{
    const int row  = blockIdx.x * 8 + (threadIdx.x >> 5);   // 0 .. BN*TI-1
    const int lane = threadIdx.x & 31;
    const float* v = vin + (size_t)row * HN;
    float vals[12];
    float s = 0.f, s2 = 0.f;
#pragma unroll
    for (int j = 0; j < 12; ++j) {
        float x = v[lane + 32 * j];
        vals[j] = x; s += x; s2 += x * x;
    }
#pragma unroll
    for (int off = 16; off; off >>= 1) {
        s  += __shfl_xor(s,  off, 32);
        s2 += __shfl_xor(s2, off, 32);
    }
    const float mu  = s * (1.f / HN);
    const float var = s2 * (1.f / HN) - mu * mu;
    const float inv = rsqrtf(var + LN_EPS);
    const int b = row / TI, t = row % TI;
    _Float16* orow = outp + ((size_t)b * TP + t + 5) * HN;
#pragma unroll
    for (int j = 0; j < 12; ++j) {
        int e = lane + 32 * j;
        float y = (vals[j] - mu) * inv * gamma[e] + beta[e];
        orow[e] = (_Float16)fmaxf(y, 0.f);
    }
}

// ---------------------------------------------------------------------------
// LayerNorm + ReLU + linear(HN->1) + token mask -> predicted durations output
// ---------------------------------------------------------------------------
__global__ __launch_bounds__(256) void ln_relu_lin_kernel(
        const float* __restrict__ vin,     // (BN, TI, HN)
        const float* __restrict__ gamma,
        const float* __restrict__ beta,
        const float* __restrict__ lin_w,   // (HN)
        const float* __restrict__ lin_b,   // (1)
        const int*   __restrict__ token_nums,
        float*       __restrict__ outp)    // (BN, TI)
{
    const int row  = blockIdx.x * 8 + (threadIdx.x >> 5);
    const int lane = threadIdx.x & 31;
    const float* v = vin + (size_t)row * HN;
    float vals[12];
    float s = 0.f, s2 = 0.f;
#pragma unroll
    for (int j = 0; j < 12; ++j) {
        float x = v[lane + 32 * j];
        vals[j] = x; s += x; s2 += x * x;
    }
#pragma unroll
    for (int off = 16; off; off >>= 1) {
        s  += __shfl_xor(s,  off, 32);
        s2 += __shfl_xor(s2, off, 32);
    }
    const float mu  = s * (1.f / HN);
    const float var = s2 * (1.f / HN) - mu * mu;
    const float inv = rsqrtf(var + LN_EPS);
    float dot = 0.f;
#pragma unroll
    for (int j = 0; j < 12; ++j) {
        int e = lane + 32 * j;
        float y = (vals[j] - mu) * inv * gamma[e] + beta[e];
        dot += fmaxf(y, 0.f) * lin_w[e];
    }
#pragma unroll
    for (int off = 16; off; off >>= 1) dot += __shfl_xor(dot, off, 32);
    if (lane == 0) {
        const int b = row / TI, t = row % TI;
        float mv = dot + lin_b[0];
        outp[row] = (t < token_nums[b]) ? mv : 0.f;
    }
}

// ---------------------------------------------------------------------------
// Per-batch inclusive scan of durations, clipped at gt. 1024 threads / batch.
// ---------------------------------------------------------------------------
__global__ __launch_bounds__(1024) void scan_cum_kernel(
        const float* __restrict__ dur,    // (BN, TI)
        const int*   __restrict__ gt,     // (BN)
        float*       __restrict__ cum)    // (BN, TI)
{
    __shared__ float sm[2][TI];
    const int b = blockIdx.x;
    const int t = threadIdx.x;
    sm[0][t] = dur[b * TI + t];
    __syncthreads();
    int src = 0;
    for (int off = 1; off < TI; off <<= 1) {
        float v = sm[src][t];
        if (t >= off) v += sm[src][t - off];
        sm[src ^ 1][t] = v;
        __syncthreads();
        src ^= 1;
    }
    const float g = (float)gt[b];
    cum[b * TI + t] = fminf(sm[src][t], g);
}

// ---------------------------------------------------------------------------
// Length regulation: searchsorted(cum, f, 'right') + gather. Wave per frame.
// grid = (BN * TO/64), block = 256 (8 waves x 8 frames each)
// ---------------------------------------------------------------------------
__global__ __launch_bounds__(256) void gather_kernel(
        const float* __restrict__ x,      // (BN, TI, HN)
        const float* __restrict__ cum,    // (BN, TI)
        const int*   __restrict__ gt,     // (BN)
        float*       __restrict__ out)    // (BN, TO, HN)
{
    __shared__ float cs[TI];
    const int blk  = blockIdx.x;
    const int b    = blk / (TO / 64);
    const int f0   = (blk % (TO / 64)) * 64;
    for (int i = threadIdx.x; i < TI; i += 256) cs[i] = cum[b * TI + i];
    __syncthreads();
    const int lane = threadIdx.x & 31;
    const int wave = threadIdx.x >> 5;
    const float g  = (float)gt[b];
#pragma unroll
    for (int j = 0; j < 8; ++j) {
        const int   f  = f0 + wave * 8 + j;
        const float ff = (float)f;
        int lo = 0, hi = TI;
        while (lo < hi) {                 // upper_bound: first cs[idx] > ff
            int mid = (lo + hi) >> 1;
            if (cs[mid] <= ff) lo = mid + 1; else hi = mid;
        }
        const int idx   = (lo < TI - 1) ? lo : (TI - 1);
        const bool valid = ff < g;
        const float4* src = (const float4*)(x + ((size_t)b * TI + idx) * HN);
        float4*       dst = (float4*)(out + ((size_t)b * TO + f) * HN);
        const float4 z = {0.f, 0.f, 0.f, 0.f};
#pragma unroll
        for (int q = 0; q < 3; ++q) {
            float4 v = valid ? src[lane + 32 * q] : z;
            dst[lane + 32 * q] = v;
        }
    }
}

// ---------------------------------------------------------------------------
extern "C" void kernel_launch(void* const* d_in, const int* in_sizes, int n_in,
                              void* d_out, int out_size, void* d_ws, size_t ws_size,
                              hipStream_t stream) {
    const float* x        = (const float*)d_in[0];   // (BN, TI, HN)
    const int*   tok_num  = (const int*)  d_in[1];   // (BN)
    const float* dur      = (const float*)d_in[2];   // (BN, TI)
    const int*   gt       = (const int*)  d_in[3];   // (BN)
    // d_in[4] = t_out scalar (constant TO)
    const float* conv1_w  = (const float*)d_in[5];
    const float* conv1_b  = (const float*)d_in[6];
    const float* ln1_g    = (const float*)d_in[7];
    const float* ln1_b    = (const float*)d_in[8];
    const float* conv2_w  = (const float*)d_in[9];
    const float* conv2_b  = (const float*)d_in[10];
    const float* ln2_g    = (const float*)d_in[11];
    const float* ln2_b    = (const float*)d_in[12];
    const float* lin_w    = (const float*)d_in[13];
    const float* lin_b    = (const float*)d_in[14];

    float* out_x = (float*)d_out;                                   // (BN,TO,HN)
    float* out_m = (float*)d_out + (size_t)BN * TO * HN;            // (BN,TI)

    // workspace layout (all 256B-aligned sizes)
    char* ws = (char*)d_ws;
    const size_t sz_xh  = (size_t)BN * TP * HN * sizeof(_Float16);  // 25.4 MB
    const size_t sz_w   = (size_t)HN * KD * sizeof(_Float16);       //  3.2 MB
    const size_t sz_co  = (size_t)BN * TI * HN * sizeof(float);     // 50.3 MB
    _Float16* xh   = (_Float16*)(ws);
    _Float16* w1p  = (_Float16*)(ws + sz_xh);
    _Float16* w2p  = (_Float16*)(ws + sz_xh + sz_w);
    _Float16* mh   = (_Float16*)(ws + sz_xh + 2 * sz_w);
    float*    cvo  = (float*)   (ws + 2 * sz_xh + 2 * sz_w);
    float*    cum  = (float*)   (ws + 2 * sz_xh + 2 * sz_w + sz_co);

    // --- pack / pad ---
    pack_x_kernel<<<(BN * TI * HN + 255) / 256, 256, 0, stream>>>(x, xh);
    pack_w_kernel<<<(HN * HN * KW + 255) / 256, 256, 0, stream>>>(conv1_w, w1p);
    pack_w_kernel<<<(HN * HN * KW + 255) / 256, 256, 0, stream>>>(conv2_w, w2p);
    zero_pads_kernel<<<(2 * BN * 10 * HN + 255) / 256, 256, 0, stream>>>(xh, mh);

    // --- length regulator (independent of conv chain) ---
    scan_cum_kernel<<<BN, 1024, 0, stream>>>(dur, gt, cum);
    gather_kernel<<<BN * (TO / 64), 256, 0, stream>>>(x, cum, gt, out_x);

    // --- duration predictor ---
    dim3 cgrid(BN * TI / 64, HN / 64);
    conv_wmma_kernel<<<cgrid, 128, 0, stream>>>(xh, w1p, conv1_b, cvo);
    ln_relu_f16_kernel<<<BN * TI / 8, 256, 0, stream>>>(cvo, ln1_g, ln1_b, mh);
    conv_wmma_kernel<<<cgrid, 128, 0, stream>>>(mh, w2p, conv2_b, cvo);
    ln_relu_lin_kernel<<<BN * TI / 8, 256, 0, stream>>>(cvo, ln2_g, ln2_b,
                                                        lin_w, lin_b, tok_num, out_m);
}
